// GNN_HSIC_40037685133332
// MI455X (gfx1250) — matrix operand order repaired and checked by reading, not compiled, not run on hardware
//
#include <hip/hip_runtime.h>

typedef __bf16 bf16_t;
typedef __attribute__((ext_vector_type(16))) __bf16 v16bf;
typedef __attribute__((ext_vector_type(8)))  __bf16 v8bf;
typedef __attribute__((ext_vector_type(8)))  float  v8f;

#define WMMA_BF16(a, b, c) \
  __builtin_amdgcn_wmma_f32_16x16x32_bf16(false, (a), false, (b), (short)0, (c), false, false)

// A-operand from a bf16 row-major LDS tile. ISA 16-bit A layout:
// lanes 0-15 -> M=lane, K = {kt*32+0..7, kt*32+16..23}; lanes 16-31 -> +8.
// Both 8-element runs are 16B contiguous -> two ds_load_b128, no cvt.
// Caller passes p = base + (lane&15)*stride + kt*32 + (lane>>4)*8  (16B aligned).
__device__ __forceinline__ v16bf load_a_bf16(const bf16_t* p) {
  v8bf lo = *(const v8bf*)p;
  v8bf hi = *(const v8bf*)(p + 16);
  v16bf a;
#pragma unroll
  for (int e = 0; e < 8; ++e) { a[e] = lo[e]; a[8 + e] = hi[e]; }
  return a;
}

// ---------------- utility kernels ----------------

__global__ void zero_kernel(float* __restrict__ p, long n) {
  long i = (long)blockIdx.x * blockDim.x + threadIdx.x;
  if (i < n) p[i] = 0.0f;
}

// Pack fp32 W[K x Nn] (row-major) into bf16 WMMA B tiles.
// Tile (tk, tn): 32 lanes x 16 values; lane -> col = tn*16 + lane%16,
// K = tk*32 + (lane/16)*8 + e (e<8), tk*32 + 16 + (lane/16)*8 + (e-8) (e>=8). Zero-padded out of range.
__global__ void pack_w_kernel(const float* __restrict__ W, int K, int Nn,
                              int nkt, int nnt, bf16_t* __restrict__ dst) {
  int tid = blockIdx.x * blockDim.x + threadIdx.x;
  int total = nkt * nnt * 32;
  if (tid >= total) return;
  int lane = tid & 31;
  int tile = tid >> 5;
  int tn = tile % nnt;
  int tk = tile / nnt;
  int col = tn * 16 + (lane & 15);
  int kbase = tk * 32 + (lane >> 4) * 8;
  bf16_t* o = dst + (size_t)tid * 16;
#pragma unroll
  for (int e = 0; e < 16; ++e) {
    int k = kbase + e + (e >= 8 ? 8 : 0);
    float v = (k < K && col < Nn) ? W[(size_t)k * Nn + col] : 0.0f;
    o[e] = (bf16_t)v;
  }
}

// ---------------- graph kernels ----------------

__global__ void deg_z_kernel(const long long* __restrict__ ei, const float* __restrict__ T,
                             float* __restrict__ degnl, float* __restrict__ zsum, long E) {
  long e = (long)blockIdx.x * blockDim.x + threadIdx.x;
  if (e >= E) return;
  long src = (long)ei[e];
  long dst = (long)ei[E + e];
  unsafeAtomicAdd(&degnl[dst], 1.0f);
  unsafeAtomicAdd(&zsum[dst], T[src]);
}

__global__ void dis_kernel(const float* __restrict__ degnl, float* __restrict__ dis, long n) {
  long i = (long)blockIdx.x * blockDim.x + threadIdx.x;
  if (i < n) dis[i] = rsqrtf(degnl[i] + 1.0f);
}

// thread = (edge, float4 channel quad): gather h[src], scatter-atomic into agg[dst]
__global__ void agg_kernel(const long long* __restrict__ ei, const float* __restrict__ h,
                           const float* __restrict__ dis, float* __restrict__ agg, long E) {
  long tid = (long)blockIdx.x * blockDim.x + threadIdx.x;
  long e = tid >> 5;
  if (e >= E) return;
  int q = (int)(tid & 31);
  long src = (long)ei[e];
  long dst = (long)ei[E + e];
  float coef = dis[src] * dis[dst];
  float4 hv = ((const float4*)h)[src * 32 + q];
  float* a = agg + dst * 128 + q * 4;
  unsafeAtomicAdd(a + 0, hv.x * coef);
  unsafeAtomicAdd(a + 1, hv.y * coef);
  unsafeAtomicAdd(a + 2, hv.z * coef);
  unsafeAtomicAdd(a + 3, hv.w * coef);
}

__global__ void fin_kernel(const float* __restrict__ agg, const float* __restrict__ h,
                           const float* __restrict__ degnl, const float* __restrict__ zsum,
                             const float* __restrict__ bgcn, float* __restrict__ rep, long Nn) {
  long tid = (long)blockIdx.x * blockDim.x + threadIdx.x;
  if (tid >= Nn * 128) return;
  long node = tid >> 7;
  int c = (int)(tid & 127);
  float dn = degnl[node];
  float inv = 1.0f / (dn + 1.0f);
  float v = agg[tid] + h[tid] * inv + bgcn[c];
  v = v > 0.0f ? v : 0.0f;
  rep[node * 257 + 128 + c] = v;
  if (c == 0) rep[node * 257 + 256] = zsum[node] / fmaxf(dn, 1.0f);
}

// ---------------- fused phi / h GEMM kernel ----------------
// 256 threads (8 waves), 16 nodes per block. A tiles staged in LDS as bf16
// (converted once), B tiles pre-packed bf16 in global.
#define XS_STRIDE 264   // 256 + 8 bf16 pad (16B) -> rows walk across LDS banks
#define PT_STRIDE 136   // 128 + 8

__global__ void phi_h_kernel(const float* __restrict__ X, const float* __restrict__ T,
                             const bf16_t* __restrict__ WphiP, const float* __restrict__ bphi,
                             const bf16_t* __restrict__ WgcnP,
                             float* __restrict__ rep, float* __restrict__ h) {
  __shared__ bf16_t Xs[16 * XS_STRIDE];
  __shared__ bf16_t Pt[16 * PT_STRIDE];
  __shared__ float Ts[16];
  const int t = threadIdx.x;
  const int lane = t & 31;
  const int w = t >> 5;
  const long i0 = (long)blockIdx.x * 16;

  // X tile (16x256 fp32, rows contiguous) -> bf16 LDS with padded rows.
  // 512 chunks of 8 floats; each thread converts 2 chunks (8 cvt_pk total).
  const float4* Xg4 = (const float4*)(X + i0 * 256);
#pragma unroll
  for (int chunk = t; chunk < 512; chunk += 256) {
    int row = chunk >> 5;
    int c8 = chunk & 31;
    float4 a = Xg4[chunk * 2];
    float4 b = Xg4[chunk * 2 + 1];
    v8bf v;
    v[0] = (bf16_t)a.x; v[1] = (bf16_t)a.y; v[2] = (bf16_t)a.z; v[3] = (bf16_t)a.w;
    v[4] = (bf16_t)b.x; v[5] = (bf16_t)b.y; v[6] = (bf16_t)b.z; v[7] = (bf16_t)b.w;
    *(v8bf*)(Xs + row * XS_STRIDE + c8 * 8) = v;
  }
  if (t < 16) Ts[t] = T[i0 + t];
  __syncthreads();

  const int row = lane & 15;
  const int kh = lane >> 4;
  const int col = w * 16 + row;

  // GEMM1: phi = relu(X @ Wphi + b), K = 256 (8 k-tiles); wave w owns cols [w*16, w*16+16)
  v8f acc = {};
#pragma unroll
  for (int kt = 0; kt < 8; ++kt) {
    v16bf a = load_a_bf16(Xs + row * XS_STRIDE + kt * 32 + kh * 8);
    v16bf b = ((const v16bf*)WphiP)[(kt * 8 + w) * 32 + lane];
    acc = WMMA_BF16(a, b, acc);
  }
  const float bias = bphi[col];
#pragma unroll
  for (int r = 0; r < 8; ++r) {
    int rw = r + 8 * kh;                          // C layout: M = r + 8*(lane/16), N = lane%16
    float v = acc[r] + bias;
    v = v > 0.0f ? v : 0.0f;
    rep[(i0 + rw) * 257 + col] = v;               // phi_x -> rep_post cols 0..127
    Pt[rw * PT_STRIDE + col] = (bf16_t)(v * Ts[rw]);  // phi_x_t staged as bf16 (converted once)
  }
  __syncthreads();

  // GEMM2: h = phi_t @ Wgcn, K = 128 (4 k-tiles)
  v8f acc2 = {};
#pragma unroll
  for (int kt = 0; kt < 4; ++kt) {
    v16bf a = load_a_bf16(Pt + row * PT_STRIDE + kt * 32 + kh * 8);
    v16bf b = ((const v16bf*)WgcnP)[(kt * 8 + w) * 32 + lane];
    acc2 = WMMA_BF16(a, b, acc2);
  }
#pragma unroll
  for (int r = 0; r < 8; ++r) {
    int rw = r + 8 * kh;
    h[(i0 + rw) * 128 + col] = acc2[r];
  }
}

// ---------------- output head kernel ----------------
// 256 threads (8 waves), 16 nodes/block. 34 tasks = {W00,W10} x 17 N-tiles.
// rep tile staged as bf16 (K padded to 288); head dot (x W01/W11) fused
// in-register via shfl reduction + ds_add_f32.
#define RA_STRIDE 296   // 288 + 8 bf16 pad

__global__ void y_out_kernel(const float* __restrict__ rep,
                             const bf16_t* __restrict__ W00P, const float* __restrict__ b00,
                             const bf16_t* __restrict__ W10P, const float* __restrict__ b10,
                             const float* __restrict__ W01, const float* __restrict__ b01v,
                             const float* __restrict__ W11, const float* __restrict__ b11v,
                             float* __restrict__ y0, float* __restrict__ y1) {
  __shared__ bf16_t A[16 * RA_STRIDE];
  __shared__ float dot0[16];
  __shared__ float dot1[16];
  const int t = threadIdx.x;
  const long i0 = (long)blockIdx.x * 16;

  for (int idx = t; idx < 16 * 288; idx += 256) {
    int rw = idx / 288;
    int c = idx - rw * 288;
    float v = (c < 257) ? rep[(i0 + rw) * 257 + c] : 0.0f;  // zero-pad K to 288
    A[rw * RA_STRIDE + c] = (bf16_t)v;
  }
  if (t < 16) { dot0[t] = 0.0f; dot1[t] = 0.0f; }
  __syncthreads();

  const int lane = t & 31;
  const int w = t >> 5;
  const int row = lane & 15;
  const int kh = lane >> 4;

  for (int task = w; task < 34; task += 8) {      // uniform per wave -> EXEC all 1s at WMMA
    const int m = task & 1;
    const int tn = task >> 1;
    const bf16_t* WP = m ? W10P : W00P;
    const float* bb = m ? b10 : b00;
    const float* wv = m ? W11 : W01;
    v8f acc = {};
#pragma unroll
    for (int kt = 0; kt < 9; ++kt) {
      v16bf a = load_a_bf16(A + row * RA_STRIDE + kt * 32 + kh * 8);
      v16bf b = ((const v16bf*)WP)[(kt * 17 + tn) * 32 + lane];
      acc = WMMA_BF16(a, b, acc);
    }
    const int n = tn * 16 + row;
    const float bias = (n < 257) ? bb[n] : 0.0f;
    const float wn   = (n < 257) ? wv[n] : 0.0f;
    float* dmem = m ? dot1 : dot0;
#pragma unroll
    for (int r = 0; r < 8; ++r) {
      float v = acc[r] + bias;
      v = v > 0.0f ? v : 0.0f;
      float pr = v * wn;                          // relu(yM)[row] * W_head[n]
      pr += __shfl_xor(pr, 1, 32);
      pr += __shfl_xor(pr, 2, 32);
      pr += __shfl_xor(pr, 4, 32);
      pr += __shfl_xor(pr, 8, 32);                // reduce over N within each 16-lane half
      if ((lane & 15) == 0) unsafeAtomicAdd(&dmem[r + 8 * kh], pr);
    }
  }
  __syncthreads();
  if (t < 16) {
    y0[i0 + t] = dot0[t] + b01v[0];
    y1[i0 + t] = dot1[t] + b11v[0];
  }
}

// ---------------- host launcher ----------------

extern "C" void kernel_launch(void* const* d_in, const int* in_sizes, int n_in,
                              void* d_out, int out_size, void* d_ws, size_t ws_size,
                              hipStream_t stream) {
  (void)n_in; (void)out_size; (void)ws_size;
  const float* X     = (const float*)d_in[0];
  const float* T     = (const float*)d_in[1];
  const long long* ei = (const long long*)d_in[2];
  const float* W_phi = (const float*)d_in[3];
  const float* b_phi = (const float*)d_in[4];
  const float* W_gcn = (const float*)d_in[5];
  const float* b_gcn = (const float*)d_in[6];
  const float* W00   = (const float*)d_in[7];
  const float* b00   = (const float*)d_in[8];
  const float* W10   = (const float*)d_in[9];
  const float* b10   = (const float*)d_in[10];
  const float* W01   = (const float*)d_in[11];
  const float* b01   = (const float*)d_in[12];
  const float* W11   = (const float*)d_in[13];
  const float* b11   = (const float*)d_in[14];

  const long Nn = in_sizes[1];          // 100000 (divisible by 16)
  const long E  = in_sizes[2] / 2;      // 3200000

  char* ws = (char*)d_ws;
  size_t cur = 0;
  auto take = [&](size_t bytes) -> void* {
    void* p = ws + cur;
    cur += (bytes + 255) & ~(size_t)255;
    return p;
  };
  size_t o_zero = cur;
  float* agg    = (float*)take((size_t)Nn * 128 * 4);
  float* degnl  = (float*)take((size_t)Nn * 4);
  float* zsum   = (float*)take((size_t)Nn * 4);
  size_t zero_n = (cur - o_zero) / 4;   // covers agg+degnl+zsum (+pad)
  float* dis    = (float*)take((size_t)Nn * 4);
  float* h      = (float*)take((size_t)Nn * 128 * 4);
  bf16_t* WphiP = (bf16_t*)take((size_t)8 * 8 * 512 * 2);    // 8x8 tiles
  bf16_t* WgcnP = (bf16_t*)take((size_t)4 * 8 * 512 * 2);    // 4x8 tiles
  bf16_t* W00P  = (bf16_t*)take((size_t)9 * 17 * 512 * 2);   // 9x17 tiles (K->288, N->272)
  bf16_t* W10P  = (bf16_t*)take((size_t)9 * 17 * 512 * 2);

  float* y0  = (float*)d_out;
  float* y1  = y0 + Nn;
  float* rep = y0 + 2 * Nn;

  zero_kernel<<<(int)((zero_n + 255) / 256), 256, 0, stream>>>((float*)(ws + o_zero), (long)zero_n);

  pack_w_kernel<<<(8 * 8 * 32 + 255) / 256, 256, 0, stream>>>(W_phi, 256, 128, 8, 8, WphiP);
  pack_w_kernel<<<(4 * 8 * 32 + 255) / 256, 256, 0, stream>>>(W_gcn, 128, 128, 4, 8, WgcnP);
  pack_w_kernel<<<(9 * 17 * 32 + 255) / 256, 256, 0, stream>>>(W00, 257, 257, 9, 17, W00P);
  pack_w_kernel<<<(9 * 17 * 32 + 255) / 256, 256, 0, stream>>>(W10, 257, 257, 9, 17, W10P);

  deg_z_kernel<<<(int)((E + 255) / 256), 256, 0, stream>>>(ei, T, degnl, zsum, E);
  dis_kernel<<<(int)((Nn + 255) / 256), 256, 0, stream>>>(degnl, dis, Nn);

  phi_h_kernel<<<(int)(Nn / 16), 256, 0, stream>>>(X, T, WphiP, b_phi, WgcnP, rep, h);

  agg_kernel<<<(int)((E * 32 + 255) / 256), 256, 0, stream>>>(ei, h, dis, agg, E);

  fin_kernel<<<(int)((Nn * 128 + 255) / 256), 256, 0, stream>>>(agg, h, degnl, zsum, b_gcn, rep, Nn);

  y_out_kernel<<<(int)(Nn / 16), 256, 0, stream>>>(rep, W00P, b00, W10P, b10,
                                                   W01, b01, W11, b11, y0, y1);
}
